// SparseAutoencoder_20048907337838
// MI455X (gfx1250) — matrix operand order repaired and checked by reading, not compiled, not run on hardware
//
#include <hip/hip_runtime.h>

// Problem constants (from reference)
#define BATCH 16384
#define DIMD  768
#define DIML  16384
#define TOPK  32

// ---------------- WMMA types ----------------
typedef __attribute__((ext_vector_type(16))) __bf16 v16bf;
typedef __attribute__((ext_vector_type(8)))  float  v8f;

union FragBF { unsigned int u[8]; v16bf v; };

__device__ __forceinline__ unsigned int pack_bf16(float a, float b) {
    unsigned int ua = __float_as_uint(a);
    unsigned int ub = __float_as_uint(b);
    ua += 0x7FFFu + ((ua >> 16) & 1u);   // round-to-nearest-even
    ub += 0x7FFFu + ((ub >> 16) & 1u);
    return (ua >> 16) | (ub & 0xFFFF0000u);
}

// =====================================================================
// Kernel 1: encode GEMM  h[b,l] = sum_d x[b,d] * W_enc[l,d]   (bf16 WMMA)
// Block tile 128x128, K-slab 32, ping-pong LDS double buffering.
// 8 waves: waveRow(0..3) x waveCol(0..1); each wave owns a 32x64 tile
// = 2x4 WMMA 16x16x32 accumulators.
// =====================================================================
__global__ __launch_bounds__(256) void encode_kernel(
        const float* __restrict__ x,
        const float* __restrict__ We,
        float* __restrict__ h) {
    __shared__ unsigned int As[2][128][17];   // 128 rows x 32 bf16 (16 uints) + pad
    __shared__ unsigned int Bs[2][128][17];

    const int t       = threadIdx.x;
    const int wave    = t >> 5;
    const int lane    = t & 31;
    const int waveRow = wave & 3;          // 32 rows each
    const int waveCol = wave >> 2;         // 64 cols each
    const int mBase   = blockIdx.y * 128;  // rows of x
    const int nBase   = blockIdx.x * 128;  // rows of W_enc (= cols of h)

    const int l16   = lane & 15;
    const int khalf = lane >> 4;

    v8f acc[2][4];
    const v8f zf = {};
#pragma unroll
    for (int i = 0; i < 2; ++i)
#pragma unroll
        for (int j = 0; j < 4; ++j) acc[i][j] = zf;

    // cooperative-load mapping: 256 threads cover a 128x32 fp32 tile
    // (one float4 each, 4 row passes)
    const int lr  = t >> 3;        // 0..31
    const int lc  = (t & 7) * 4;   // 0,4,...,28
    const int lcu = lc >> 1;       // uint column in LDS

    // ---- prologue: fill buffer 0 with slab kt=0 ----
#pragma unroll
    for (int rr = 0; rr < 4; ++rr) {
        const int r = lr + rr * 32;
        const float4 xa = *reinterpret_cast<const float4*>(
            &x[(size_t)(mBase + r) * DIMD + lc]);
        As[0][r][lcu]     = pack_bf16(xa.x, xa.y);
        As[0][r][lcu + 1] = pack_bf16(xa.z, xa.w);
        const float4 wb = *reinterpret_cast<const float4*>(
            &We[(size_t)(nBase + r) * DIMD + lc]);
        Bs[0][r][lcu]     = pack_bf16(wb.x, wb.y);
        Bs[0][r][lcu + 1] = pack_bf16(wb.z, wb.w);
    }
    __syncthreads();

    int buf = 0;
    for (int kt = 0; kt < DIMD; kt += 32) {
        const bool more = (kt + 32) < DIMD;

        // ---- issue global loads for slab kt+32 (kept in VGPRs; loadcnt in flight) ----
        float4 nxa[4], nwb[4];
        if (more) {
#pragma unroll
            for (int rr = 0; rr < 4; ++rr) {
                const int r = lr + rr * 32;
                nxa[rr] = *reinterpret_cast<const float4*>(
                    &x[(size_t)(mBase + r) * DIMD + kt + 32 + lc]);
                nwb[rr] = *reinterpret_cast<const float4*>(
                    &We[(size_t)(nBase + r) * DIMD + kt + 32 + lc]);
            }
            if (kt + 64 < DIMD) {  // warm L2 for slab kt+64 (global_prefetch_b8)
                __builtin_prefetch(&x[(size_t)(mBase + lr) * DIMD + kt + 64 + lc], 0, 3);
                __builtin_prefetch(&We[(size_t)(nBase + lr) * DIMD + kt + 64 + lc], 0, 3);
            }
        }

        // ---- compute on buffer `buf` (ds loads + WMMA overlap the global fetch) ----
        FragBF fa[2];
#pragma unroll
        for (int rt = 0; rt < 2; ++rt) {
            const int rowi = waveRow * 32 + rt * 16 + l16;
#pragma unroll
            for (int p = 0; p < 4; ++p) {
                fa[rt].u[p]     = As[buf][rowi][khalf * 4 + p];     // K = khalf*8 + 2p
                fa[rt].u[4 + p] = As[buf][rowi][8 + khalf * 4 + p]; // K = 16 + khalf*8 + 2p
            }
        }
        FragBF fb[4];
#pragma unroll
        for (int ct = 0; ct < 4; ++ct) {
            const int coli = waveCol * 64 + ct * 16 + l16;
#pragma unroll
            for (int p = 0; p < 4; ++p) {
                fb[ct].u[p]     = Bs[buf][coli][khalf * 4 + p];     // K = khalf*8 + 2p
                fb[ct].u[4 + p] = Bs[buf][coli][8 + khalf * 4 + p]; // K = 16 + khalf*8 + 2p
            }
        }
#pragma unroll
        for (int rt = 0; rt < 2; ++rt)
#pragma unroll
            for (int ct = 0; ct < 4; ++ct)
                acc[rt][ct] = __builtin_amdgcn_wmma_f32_16x16x32_bf16(
                    false, fa[rt].v, false, fb[ct].v,
                    (short)0, acc[rt][ct], false, false);

        // ---- convert + store slab kt+32 into the other buffer ----
        if (more) {
            const int nb = buf ^ 1;
#pragma unroll
            for (int rr = 0; rr < 4; ++rr) {
                const int r = lr + rr * 32;
                As[nb][r][lcu]     = pack_bf16(nxa[rr].x, nxa[rr].y);
                As[nb][r][lcu + 1] = pack_bf16(nxa[rr].z, nxa[rr].w);
                Bs[nb][r][lcu]     = pack_bf16(nwb[rr].x, nwb[rr].y);
                Bs[nb][r][lcu + 1] = pack_bf16(nwb[rr].z, nwb[rr].w);
            }
        }
        __syncthreads();   // single barrier per K-slab
        buf ^= 1;
    }

    // C layout: VGPR i -> M = i (+8 for lanes 16..31), N = lane&15
#pragma unroll
    for (int rt = 0; rt < 2; ++rt) {
#pragma unroll
        for (int ct = 0; ct < 4; ++ct) {
            const int m0 = mBase + waveRow * 32 + rt * 16 + khalf * 8;
            const int n0 = nBase + waveCol * 64 + ct * 16 + l16;
#pragma unroll
            for (int i = 0; i < 8; ++i)
                h[(size_t)(m0 + i) * DIML + n0] = acc[rt][ct][i];
        }
    }
}

// =====================================================================
// Kernel 2: per-row top-32 by |value|, in place; record (idx,val) to ws.
// One 256-thread block per row; 64 values per thread held in registers.
// Two barriers per selection round.
// =====================================================================
__global__ __launch_bounds__(256) void topk_kernel(
        float* __restrict__ a,
        float* __restrict__ wsVal,
        int*   __restrict__ wsIdx) {
    const int row = blockIdx.x;
    const int t   = threadIdx.x;
    float* arow = a + (size_t)row * DIML;

    float v[64];
#pragma unroll
    for (int i = 0; i < 64; ++i) v[i] = arow[t + (i << 8)];

    unsigned long long selMask = 0ull;

    __shared__ float        sAbs[8];
    __shared__ unsigned int sIdx[8];
    __shared__ float        sVal[8];

    for (int it = 0; it < TOPK; ++it) {
        float bestA = -1.0f, bestV = 0.0f;
        unsigned int bestI = 0xFFFFFFFFu;
#pragma unroll
        for (int i = 0; i < 64; ++i) {
            const float av = ((selMask >> i) & 1ull) ? -1.0f : fabsf(v[i]);
            const unsigned int gi = (unsigned)(t + (i << 8));
            if (av > bestA || (av == bestA && gi < bestI)) {
                bestA = av; bestI = gi; bestV = v[i];
            }
        }
        // wave32 reduction
        for (int off = 16; off > 0; off >>= 1) {
            const float        oa = __shfl_xor(bestA, off, 32);
            const unsigned int oi = __shfl_xor(bestI, off, 32);
            const float        ov = __shfl_xor(bestV, off, 32);
            if (oa > bestA || (oa == bestA && oi < bestI)) {
                bestA = oa; bestI = oi; bestV = ov;
            }
        }
        if ((t & 31) == 0) { sAbs[t >> 5] = bestA; sIdx[t >> 5] = bestI; sVal[t >> 5] = bestV; }
        __syncthreads();

        // every thread redundantly reduces the 8 wave winners (deterministic)
        float ba = sAbs[0], bv = sVal[0];
        unsigned int bi = sIdx[0];
#pragma unroll
        for (int w = 1; w < 8; ++w) {
            if (sAbs[w] > ba || (sAbs[w] == ba && sIdx[w] < bi)) {
                ba = sAbs[w]; bi = sIdx[w]; bv = sVal[w];
            }
        }
        if (t == 0) {
            wsVal[(size_t)row * TOPK + it] = bv;
            wsIdx[(size_t)row * TOPK + it] = (int)bi;
        }
        if ((bi & 255u) == (unsigned)t)
            selMask |= (1ull << (bi >> 8));
        __syncthreads();   // protect sAbs/sIdx/sVal reuse next round
    }

#pragma unroll
    for (int i = 0; i < 64; ++i)
        arow[t + (i << 8)] = ((selMask >> i) & 1ull) ? v[i] : 0.0f;
}

// =====================================================================
// Kernel 3: transpose W_dec [D,L] -> Wt [L,D] (LDS 32x32 tiles)
// =====================================================================
__global__ __launch_bounds__(256) void transpose_kernel(
        const float* __restrict__ Wd, float* __restrict__ Wt) {
    __shared__ float tile[32][33];
    const int lb = blockIdx.x * 32;   // L base
    const int db = blockIdx.y * 32;   // D base
    const int tx = threadIdx.x, ty = threadIdx.y;  // 32 x 8
#pragma unroll
    for (int i = 0; i < 4; ++i)
        tile[ty + i * 8][tx] = Wd[(size_t)(db + ty + i * 8) * DIML + lb + tx];
    __syncthreads();
#pragma unroll
    for (int i = 0; i < 4; ++i)
        Wt[(size_t)(lb + ty + i * 8) * DIMD + db + tx] = tile[tx][ty + i * 8];
}

// =====================================================================
// Kernel 4: sparse decode  recon[b,d] = sum_k val[b,k] * Wt[idx[b,k], d]
// =====================================================================
__global__ __launch_bounds__(256) void decode_kernel(
        const float* __restrict__ wsVal,
        const int*   __restrict__ wsIdx,
        const float* __restrict__ Wt,
        float* __restrict__ recon) {
    const int row = blockIdx.x;
    const int t   = threadIdx.x;
    __shared__ float sv[TOPK];
    __shared__ int   si[TOPK];
    if (t < TOPK) {
        sv[t] = wsVal[(size_t)row * TOPK + t];
        si[t] = wsIdx[(size_t)row * TOPK + t];
    }
    __syncthreads();
    float acc0 = 0.f, acc1 = 0.f, acc2 = 0.f;
#pragma unroll 4
    for (int k = 0; k < TOPK; ++k) {
        const float* wr = Wt + (size_t)si[k] * DIMD;
        const float s = sv[k];
        acc0 += s * wr[t];
        acc1 += s * wr[t + 256];
        acc2 += s * wr[t + 512];
    }
    float* r = recon + (size_t)row * DIMD;
    r[t] = acc0; r[t + 256] = acc1; r[t + 512] = acc2;
}

// =====================================================================
extern "C" void kernel_launch(void* const* d_in, const int* in_sizes, int n_in,
                              void* d_out, int out_size, void* d_ws, size_t ws_size,
                              hipStream_t stream) {
    (void)in_sizes; (void)n_in; (void)out_size; (void)ws_size;

    const float* x     = (const float*)d_in[0];   // [B, D]
    const float* W_enc = (const float*)d_in[1];   // [L, D]
    const float* W_dec = (const float*)d_in[2];   // [D, L]

    float* recon = (float*)d_out;                              // [B, D]
    float* a     = (float*)d_out + (size_t)BATCH * DIMD;       // [B, L]

    float* Wt    = (float*)d_ws;                               // [L, D] transposed decoder
    float* wsVal = Wt + (size_t)DIML * DIMD;                   // [B, 32]
    int*   wsIdx = (int*)(wsVal + (size_t)BATCH * TOPK);       // [B, 32]

    // 1) encode GEMM (bf16 WMMA), h written straight into the `a` output slot
    encode_kernel<<<dim3(DIML / 128, BATCH / 128), 256, 0, stream>>>(x, W_enc, a);

    // 2) top-32 by |value| per row (in place), record (idx, val)
    topk_kernel<<<BATCH, 256, 0, stream>>>(a, wsVal, wsIdx);

    // 3) transpose W_dec for coalesced sparse gathers
    transpose_kernel<<<dim3(DIML / 32, DIMD / 32), dim3(32, 8), 0, stream>>>(W_dec, Wt);

    // 4) sparse decode
    decode_kernel<<<BATCH, 256, 0, stream>>>(wsVal, wsIdx, Wt, recon);
}
